// SheafHyperGCN_31842887533298
// MI455X (gfx1250) — matrix operand order repaired and checked by reading, not compiled, not run on hardware
//
#include <hip/hip_runtime.h>
#include <math.h>

#define N_NODES   100000
#define N_EDGES   50000
#define KNODES    8
#define F_IN      512
#define HID       64
#define D_STALK   6
#define C_MID     16
#define C_OUT     32
#define HD_DIM    (HID * D_STALK)          /* 384 */
#define M_ALL     (N_NODES + N_EDGES)      /* 150000 rows through input GEMM */
#define NROWS     (N_NODES * D_STALK)      /* 600000 = rows of Hd */

typedef __attribute__((ext_vector_type(16))) __bf16        v16bf;
typedef __attribute__((ext_vector_type(8)))  float         v8f;
typedef __attribute__((ext_vector_type(4)))  unsigned int  v4u;
typedef __attribute__((ext_vector_type(8)))  int           v8i;
typedef __attribute__((ext_vector_type(4)))  int           v4i;

union frag16  { unsigned int u[8]; v16bf v; };        /* 8 packed bf16 pairs */
union frag16q { v4u q[2];          v16bf v; };        /* same, as 2x128-bit  */

/* ---------- helpers ---------- */
__device__ __forceinline__ unsigned short f2bfbits(float f) {
  unsigned u = __float_as_uint(f);
  unsigned r = u + 0x7FFFu + ((u >> 16) & 1u);   /* round-to-nearest-even */
  return (unsigned short)(r >> 16);
}
__device__ __forceinline__ unsigned int pk2(float lo, float hi) {
  return (unsigned)f2bfbits(lo) | ((unsigned)f2bfbits(hi) << 16);
}

/* ---------- packed bf16 conversion passes ---------- */
__global__ __launch_bounds__(256) void k_cvt_x(
    const float* __restrict__ x, const float* __restrict__ ea,
    unsigned int* __restrict__ xb) {
  const long i = (long)blockIdx.x * blockDim.x + threadIdx.x;
  if (i >= (long)M_ALL * (F_IN / 2)) return;
  const long row = i / (F_IN / 2);
  const int  pr  = (int)(i % (F_IN / 2));
  const float* src = (row < N_NODES) ? (x + row * F_IN)
                                     : (ea + (row - N_NODES) * F_IN);
  const float2 f = *(const float2*)(src + 2 * pr);
  xb[i] = pk2(f.x, f.y);
}

/* Wt[n, k] = W_lin[k, n] as packed bf16 (row n holds 512 k, 256 uints) */
__global__ __launch_bounds__(256) void k_cvt_wt(
    const float* __restrict__ W, unsigned int* __restrict__ Wt) {
  const int i = blockIdx.x * blockDim.x + threadIdx.x;
  if (i >= HD_DIM * (F_IN / 2)) return;
  const int n = i / (F_IN / 2);
  const int k = 2 * (i % (F_IN / 2));
  Wt[i] = pk2(W[(long)k * HD_DIM + n], W[(long)(k + 1) * HD_DIM + n]);
}

/* transposed packed bf16 copies of the two small layer weights */
__global__ __launch_bounds__(512) void k_cvt_w01(
    const float* __restrict__ W0, const float* __restrict__ W1,
    unsigned int* __restrict__ W0t, unsigned int* __restrict__ W1t) {
  const int i = blockIdx.x * blockDim.x + threadIdx.x;
  if (i < C_MID * (HID / 2)) {                  /* W0t: [16][32 uints] */
    const int n = i >> 5, k = 2 * (i & 31);
    W0t[i] = pk2(W0[k * C_MID + n], W0[(k + 1) * C_MID + n]);
  }
  if (i < C_OUT * (C_MID / 2)) {                /* W1t: [32][8 uints] */
    const int n = i >> 3, k = 2 * (i & 7);
    W1t[i] = pk2(W1[k * C_OUT + n], W1[(k + 1) * C_OUT + n]);
  }
}

/* =====================================================================
 * GEMM 1: H_all[150000, 384] = cat(x, ea) @ W_lin + b_lin   (bf16 WMMA)
 * 8 waves/block, one 16-row m-tile each, 64 output cols (blockIdx.y).
 * B tiles staged in LDS by the Tensor Data Mover, double-buffered
 * (64x128 bf16 = 16 KB per buffer); DMA of chunk c+1 overlaps compute
 * of chunk c. B fragments are read back with ds_load_b128 pairs.
 * ===================================================================== */
#define KCHUNK    128
#define NCHUNKS   (F_IN / KCHUNK)          /* 4 */
#define SW_ELEMS  (64 * KCHUNK)            /* 8192 ushorts = 16 KB */

__device__ __forceinline__ void tdm_issue_wtile(
    const unsigned int* Wt, int nb, int kc, unsigned lds_byte_off) {
  const unsigned long long ga =
      (unsigned long long)Wt + ((unsigned long long)(nb * 64) * F_IN + kc) * 2ull;
  v4u g0;
  g0[0] = 1u;                                        /* count=1, user mode */
  g0[1] = lds_byte_off;                              /* lds_addr           */
  g0[2] = (unsigned)(ga & 0xFFFFFFFFull);            /* global_addr lo     */
  g0[3] = (unsigned)((ga >> 32) & 0x01FFFFFFull) | 0x80000000u; /* type=2  */
  v8i g1;
  g1[0] = 0x00010000;            /* data_size=1 (2 bytes), workgroup_mask=0 */
  g1[1] = (int)(512u << 16);     /* tensor_dim0 = 512 (lo16)                */
  g1[2] = (int)(64u << 16);      /* tensor_dim0 hi | tensor_dim1 = 64       */
  g1[3] = (int)((unsigned)KCHUNK << 16); /* tensor_dim1 hi | tile_dim0      */
  g1[4] = 64;                    /* tile_dim1 = 64, tile_dim2 = 0           */
  g1[5] = 512;                   /* tensor_dim0_stride = 512                */
  g1[6] = 0;
  g1[7] = 0;
  v4i gz = {0, 0, 0, 0};
#if defined(__clang_major__) && (__clang_major__ >= 23)
  v8i gz8 = {0, 0, 0, 0, 0, 0, 0, 0};
  __builtin_amdgcn_tensor_load_to_lds(g0, g1, gz, gz, gz8, 0);
#else
  __builtin_amdgcn_tensor_load_to_lds(g0, g1, gz, gz, 0);
#endif
}

__global__ __launch_bounds__(256) void k_gemm_in(
    const unsigned int* __restrict__ xb, const unsigned int* __restrict__ Wt,
    const float* __restrict__ bias, float* __restrict__ H,
    unsigned short* __restrict__ Hb) {
  __shared__ unsigned short sW[2][SW_ELEMS];    /* 32 KB double buffer */
  const int wave = threadIdx.x >> 5;
  const int lane = threadIdx.x & 31;
  int mt = blockIdx.x * 8 + wave;
  const bool active = (mt < M_ALL / 16);
  if (!active) mt = M_ALL / 16 - 1;             /* clamp: must keep barriers */
  const int nb   = blockIdx.y;                  /* 0..5 */
  const int m    = lane & 15;
  const int half = lane >> 4;
  const long rowc = (long)mt * 16 + m;

  v8f acc[4];
#pragma unroll
  for (int j = 0; j < 4; ++j)
#pragma unroll
    for (int r = 0; r < 8; ++r) acc[j][r] = 0.0f;

  /* prologue: DMA chunk 0 into buffer 0 */
  if (threadIdx.x == 0) tdm_issue_wtile(Wt, nb, 0, 0);

  for (int ci = 0; ci < NCHUNKS; ++ci) {
    const int kc = ci * KCHUNK;
    if (threadIdx.x == 0) {
      if (ci + 1 < NCHUNKS)
        tdm_issue_wtile(Wt, nb, kc + KCHUNK,
                        (unsigned)(((ci + 1) & 1) * SW_ELEMS * 2));
      /* steady state: 2 outstanding; <=1 means chunk ci has landed */
      if (ci + 1 < NCHUNKS) __builtin_amdgcn_s_wait_tensorcnt(1);
      else                  __builtin_amdgcn_s_wait_tensorcnt(0);
    }
    __syncthreads();
    asm volatile("" : : "r"(&sW[0][0]) : "memory");   /* TDM wrote LDS */
    /* DS byte address of this buffer (low 32 bits of the LDS aperture) */
    const unsigned ldsbase =
        (unsigned)(unsigned long long)(const void*)&sW[ci & 1][0];

#pragma unroll
    for (int kt = kc; kt < kc + KCHUNK; kt += 32) {
      frag16 a;
#pragma unroll
      for (int v = 0; v < 8; ++v) {
        const int kb = kt + (half << 4) + 2 * v;
        a.u[v] = xb[rowc * (F_IN / 2) + (kb >> 1)];
      }
#pragma unroll
      for (int j = 0; j < 4; ++j) {
        /* 8 consecutive packed uints per lane: 2x ds_load_b128 */
        const unsigned addr =
            ldsbase + (unsigned)(((j * 16 + m) * (KCHUNK / 2) +
                                  ((kt - kc) >> 1) + half * 8) * 4);
        frag16q b;
        asm volatile("ds_load_b128 %0, %2\n\t"
                     "ds_load_b128 %1, %2 offset:16\n\t"
                     "s_wait_dscnt 0"
                     : "=v"(b.q[0]), "=v"(b.q[1])
                     : "v"(addr));
        acc[j] = __builtin_amdgcn_wmma_f32_16x16x32_bf16(
            false, a.v, false, b.v, (short)0, acc[j], false, false);
      }
    }
    __syncthreads();   /* all waves done reading buf[ci&1] before reuse */
  }

  if (active) {
#pragma unroll
    for (int j = 0; j < 4; ++j)
#pragma unroll
      for (int r = 0; r < 8; ++r) {
        const int orow = mt * 16 + half * 8 + r;
        const int col  = nb * 64 + j * 16 + m;
        const float vv = acc[j][r] + bias[col];
        H[(long)orow * HD_DIM + col] = vv;
        if (orow < N_NODES) Hb[(long)orow * HD_DIM + col] = f2bfbits(vv);
      }
  }
}

/* =====================================================================
 * GEMM 2: G0[600000, 16] = Hd(bf16 view of Hb)[600000, 64] @ W0[64, 16]
 * ===================================================================== */
__global__ __launch_bounds__(256) void k_gemm_l0(
    const unsigned int* __restrict__ Hb32, const unsigned int* __restrict__ W0t,
    float* __restrict__ G0) {
  const int wave = threadIdx.x >> 5;
  const int lane = threadIdx.x & 31;
  const int mt = blockIdx.x * 8 + wave;
  if (mt >= NROWS / 16) return;
  const int m    = lane & 15;
  const int half = lane >> 4;
  const long rowc = (long)mt * 16 + m;

  v8f acc;
#pragma unroll
  for (int r = 0; r < 8; ++r) acc[r] = 0.0f;

#pragma unroll
  for (int kt = 0; kt < HID; kt += 32) {
    frag16 a, b;
#pragma unroll
    for (int v = 0; v < 8; ++v) {
      const int kb = kt + (half << 4) + 2 * v;
      a.u[v] = Hb32[rowc * (HID / 2) + (kb >> 1)];
      const int k = kt + (half << 4) + 2 * v;
      b.u[v] = W0t[m * (HID / 2) + (k >> 1)];
    }
    acc = __builtin_amdgcn_wmma_f32_16x16x32_bf16(
        false, a.v, false, b.v, (short)0, acc, false, false);
  }
#pragma unroll
  for (int r = 0; r < 8; ++r)
    G0[((long)mt * 16 + half * 8 + r) * C_MID + m] = acc[r];
}

/* =====================================================================
 * GEMM 3: G1[600000, 32] = H1(bf16)[600000, 16] @ W1[16, 32]  (K pad 32)
 * ===================================================================== */
__global__ __launch_bounds__(256) void k_gemm_l1(
    const unsigned int* __restrict__ H1b32, const unsigned int* __restrict__ W1t,
    float* __restrict__ G1) {
  const int wave = threadIdx.x >> 5;
  const int lane = threadIdx.x & 31;
  const int mt = blockIdx.x * 8 + wave;
  if (mt >= NROWS / 16) return;
  const int m    = lane & 15;
  const int half = lane >> 4;
  const long rowc = (long)mt * 16 + m;

  v8f acc[2];
#pragma unroll
  for (int j = 0; j < 2; ++j)
#pragma unroll
    for (int r = 0; r < 8; ++r) acc[j][r] = 0.0f;

  frag16 a;
#pragma unroll
  for (int v = 0; v < 8; ++v) {
    const int kb = (half << 4) + 2 * v;
    a.u[v] = (kb < C_MID) ? H1b32[rowc * (C_MID / 2) + (kb >> 1)] : 0u;
  }
#pragma unroll
  for (int j = 0; j < 2; ++j) {
    frag16 b;
#pragma unroll
    for (int v = 0; v < 8; ++v) {
      const int k = (half << 4) + 2 * v;
      b.u[v] = (k < C_MID) ? W1t[(j * 16 + m) * (C_MID / 2) + (k >> 1)] : 0u;
    }
    acc[j] = __builtin_amdgcn_wmma_f32_16x16x32_bf16(
        false, a.v, false, b.v, (short)0, acc[j], false, false);
  }
#pragma unroll
  for (int j = 0; j < 2; ++j)
#pragma unroll
    for (int r = 0; r < 8; ++r)
      G1[((long)mt * 16 + half * 8 + r) * C_OUT + j * 16 + m] = acc[j][r];
}

/* ---------- p = H @ rv (node rows only) ---------- */
__global__ __launch_bounds__(256) void k_proj(
    const float* __restrict__ H, const float* __restrict__ rv,
    float* __restrict__ p) {
  const int n = blockIdx.x * blockDim.x + threadIdx.x;
  if (n >= N_NODES) return;
  const float* h = H + (long)n * HD_DIM;
  float s = 0.0f;
  for (int k = 0; k < HD_DIM; ++k) s += h[k] * rv[k];
  p[n] = s;
}

/* ---------- node / hyperedge summaries (mean over D stalk dims) ---------- */
__global__ __launch_bounds__(256) void k_means(
    const float* __restrict__ H, float* __restrict__ xm,
    float* __restrict__ em) {
  const long i = (long)blockIdx.x * blockDim.x + threadIdx.x;
  if (i >= (long)M_ALL * HID) return;
  const long r = i / HID;
  const int  h = (int)(i % HID);
  const float* row = H + r * HD_DIM;
  float s = 0.0f;
#pragma unroll
  for (int d = 0; d < D_STALK; ++d) s += row[d * HID + h];
  s *= (1.0f / (float)D_STALK);
  if (r < N_NODES) xm[r * HID + h] = s;
  else             em[(r - N_NODES) * HID + h] = s;
}

/* ---------- per-edge argmax/argmin of projection (first-occurrence ties) */
__global__ __launch_bounds__(256) void k_argmm(
    const int* __restrict__ edge_nodes, const float* __restrict__ p,
    int* __restrict__ Se, int* __restrict__ Ie) {
  const int e = blockIdx.x * blockDim.x + threadIdx.x;
  if (e >= N_EDGES) return;
  int   bn = edge_nodes[e * KNODES];
  int   wn = bn;
  float bv = p[bn], wv = bv;
#pragma unroll
  for (int j = 1; j < KNODES; ++j) {
    const int   n = edge_nodes[e * KNODES + j];
    const float v = p[n];
    if (v > bv) { bv = v; bn = n; }
    if (v < wv) { wv = v; wn = n; }
  }
  Se[e] = bn;
  Ie[e] = wn;
}

/* ---------- sheaf MLP at the two selected incidences + vals + degrees ---- */
__device__ __forceinline__ void sheaf_row(
    const float* __restrict__ xr, const float* __restrict__ er,
    const float* __restrict__ g, const float* __restrict__ bln,
    const float* __restrict__ Ws, const float* __restrict__ bs,
    float* __restrict__ h) {
  float s = 0.0f, s2 = 0.0f;
  for (int i = 0; i < HID; ++i) { float v = xr[i]; s += v; s2 += v * v; }
  for (int i = 0; i < HID; ++i) { float v = er[i]; s += v; s2 += v * v; }
  const float mu  = s * (1.0f / 128.0f);
  const float var = s2 * (1.0f / 128.0f) - mu * mu;
  const float rs  = rsqrtf(var + 1e-5f);
  float acc[D_STALK];
#pragma unroll
  for (int d = 0; d < D_STALK; ++d) acc[d] = bs[d];
  for (int i = 0; i < 2 * HID; ++i) {
    const float v = (i < HID) ? xr[i] : er[i - HID];
    const float t = (v - mu) * rs * g[i] + bln[i];
#pragma unroll
    for (int d = 0; d < D_STALK; ++d) acc[d] += t * Ws[i * D_STALK + d];
  }
#pragma unroll
  for (int d = 0; d < D_STALK; ++d) h[d] = 1.0f / (1.0f + __expf(-acc[d]));
}

__global__ __launch_bounds__(256) void k_sheaf(
    const float* __restrict__ xm, const float* __restrict__ em,
    const float* __restrict__ ln_g, const float* __restrict__ ln_b,
    const float* __restrict__ Ws, const float* __restrict__ bs,
    const int* __restrict__ Se, const int* __restrict__ Ie,
    float* __restrict__ vals, float* __restrict__ deg) {
  const int e = blockIdx.x * blockDim.x + threadIdx.x;
  if (e >= N_EDGES) return;
  const int s = Se[e], t = Ie[e];
  const float* er = em + (long)e * HID;
  float hS[D_STALK], hI[D_STALK];
  sheaf_row(xm + (long)s * HID, er, ln_g, ln_b, Ws, bs, hS);
  sheaf_row(xm + (long)t * HID, er, ln_g, ln_b, Ws, bs, hI);
#pragma unroll
  for (int d = 0; d < D_STALK; ++d) vals[e * D_STALK + d] = hS[d] * hI[d];
  atomicAdd(&deg[s], 1.0f);     /* deg identical across stalk dims */
  atomicAdd(&deg[t], 1.0f);
}

__global__ __launch_bounds__(256) void k_dinv(float* __restrict__ deg) {
  const int n = blockIdx.x * blockDim.x + threadIdx.x;
  if (n >= N_NODES) return;
  const float d = deg[n];
  deg[n] = (d > 0.0f) ? (1.0f / d) : 0.0f;
}

/* ---------- symmetric degree-normalized scatter: agg += D^-1 A X ---------- */
__global__ __launch_bounds__(256) void k_scatter(
    const float* __restrict__ X, const float* __restrict__ vals,
    const int* __restrict__ Se, const int* __restrict__ Ie,
    const float* __restrict__ dinv, float* __restrict__ agg, int C) {
  const int i = blockIdx.x * blockDim.x + threadIdx.x;
  if (i >= N_EDGES * D_STALK) return;
  const int e = i / D_STALK;
  const int d = i % D_STALK;
  const int s = Se[e], t = Ie[e];
  const float w  = vals[i];
  const float ws = w * dinv[s];
  const float wt = w * dinv[t];
  const long rs = (long)(s * D_STALK + d) * C;
  const long rt = (long)(t * D_STALK + d) * C;
  for (int c = 0; c < C; ++c) {
    atomicAdd(&agg[rs + c], ws * X[rt + c]);
    atomicAdd(&agg[rt + c], wt * X[rs + c]);
  }
}

/* ---------- H1(bf16) = elu(G0 - agg0 + b0) ---------- */
__global__ __launch_bounds__(256) void k_elu_bf(
    const float* __restrict__ G, const float* __restrict__ agg,
    const float* __restrict__ b, unsigned short* __restrict__ out,
    int C, long total) {
  const long i = (long)blockIdx.x * blockDim.x + threadIdx.x;
  if (i >= total) return;
  const int c = (int)(i % C);
  const float v = G[i] - agg[i] + b[c];
  out[i] = f2bfbits((v > 0.0f) ? v : (__expf(v) - 1.0f));
}

/* ---------- out(f32) = elu(G - agg + b) ---------- */
__global__ __launch_bounds__(256) void k_elu_f32(
    const float* __restrict__ G, const float* __restrict__ agg,
    const float* __restrict__ b, float* __restrict__ out, int C, long total) {
  const long i = (long)blockIdx.x * blockDim.x + threadIdx.x;
  if (i >= total) return;
  const int c = (int)(i % C);
  const float v = G[i] - agg[i] + b[c];
  out[i] = (v > 0.0f) ? v : (__expf(v) - 1.0f);
}

/* ===================================================================== */
static inline char* ws_take(char** cur, size_t bytes) {
  char* p = *cur;
  *cur += (bytes + 255) & ~(size_t)255;
  return p;
}

extern "C" void kernel_launch(void* const* d_in, const int* in_sizes, int n_in,
                              void* d_out, int out_size, void* d_ws, size_t ws_size,
                              hipStream_t stream) {
  (void)in_sizes; (void)n_in; (void)out_size; (void)ws_size;
  const float* x      = (const float*)d_in[0];
  const float* ea     = (const float*)d_in[1];
  const int*   enod   = (const int*)d_in[2];
  const float* rv     = (const float*)d_in[3];
  const float* W_lin  = (const float*)d_in[4];
  const float* b_lin  = (const float*)d_in[5];
  const float* ln_g   = (const float*)d_in[6];
  const float* ln_b   = (const float*)d_in[7];
  const float* W_sh   = (const float*)d_in[8];
  const float* b_sh   = (const float*)d_in[9];
  const float* W0     = (const float*)d_in[10];
  const float* b0     = (const float*)d_in[11];
  const float* W1     = (const float*)d_in[12];
  const float* b1     = (const float*)d_in[13];
  float* out = (float*)d_out;

  char* cur = (char*)d_ws;
  unsigned int* xb   = (unsigned int*)ws_take(&cur, (size_t)M_ALL * (F_IN / 2) * 4);
  unsigned int* Wt   = (unsigned int*)ws_take(&cur, (size_t)HD_DIM * (F_IN / 2) * 4);
  unsigned int* W0t  = (unsigned int*)ws_take(&cur, (size_t)C_MID * (HID / 2) * 4);
  unsigned int* W1t  = (unsigned int*)ws_take(&cur, (size_t)C_OUT * (C_MID / 2) * 4);
  float* H_all = (float*)ws_take(&cur, (size_t)M_ALL * HD_DIM * 4);
  unsigned short* Hb = (unsigned short*)ws_take(&cur, (size_t)N_NODES * HD_DIM * 2);
  float* xm    = (float*)ws_take(&cur, (size_t)N_NODES * HID * 4);
  float* em    = (float*)ws_take(&cur, (size_t)N_EDGES * HID * 4);
  float* p     = (float*)ws_take(&cur, (size_t)N_NODES * 4);
  int*   Se    = (int*)  ws_take(&cur, (size_t)N_EDGES * 4);
  int*   Ie    = (int*)  ws_take(&cur, (size_t)N_EDGES * 4);
  float* vals  = (float*)ws_take(&cur, (size_t)N_EDGES * D_STALK * 4);
  float* deg   = (float*)ws_take(&cur, (size_t)N_NODES * 4);
  float* G0    = (float*)ws_take(&cur, (size_t)NROWS * C_MID * 4);
  float* agg0  = (float*)ws_take(&cur, (size_t)NROWS * C_MID * 4);
  unsigned short* H1b = (unsigned short*)ws_take(&cur, (size_t)NROWS * C_MID * 2);
  float* G1    = (float*)ws_take(&cur, (size_t)NROWS * C_OUT * 4);
  float* agg1  = (float*)ws_take(&cur, (size_t)NROWS * C_OUT * 4);

  hipMemsetAsync(deg,  0, (size_t)N_NODES * 4, stream);
  hipMemsetAsync(agg0, 0, (size_t)NROWS * C_MID * 4, stream);
  hipMemsetAsync(agg1, 0, (size_t)NROWS * C_OUT * 4, stream);

  /* 0. bf16 packing passes */
  k_cvt_x<<<(int)(((long)M_ALL * (F_IN / 2) + 255) / 256), 256, 0, stream>>>(x, ea, xb);
  k_cvt_wt<<<(HD_DIM * (F_IN / 2) + 255) / 256, 256, 0, stream>>>(W_lin, Wt);
  k_cvt_w01<<<1, 512, 0, stream>>>(W0, W1, W0t, W1t);

  /* 1. input GEMM: double-buffered TDM-staged B tiles + bf16 WMMA */
  k_gemm_in<<<dim3((M_ALL / 16 + 7) / 8, HD_DIM / 64), 256, 0, stream>>>(
      xb, Wt, b_lin, H_all, Hb);

  /* 2. projection + summaries */
  k_proj<<<(N_NODES + 255) / 256, 256, 0, stream>>>(H_all, rv, p);
  k_means<<<(int)(((long)M_ALL * HID + 255) / 256), 256, 0, stream>>>(H_all, xm, em);

  /* 3. edge selection + sheaf + weights + degrees */
  k_argmm<<<(N_EDGES + 255) / 256, 256, 0, stream>>>(enod, p, Se, Ie);
  k_sheaf<<<(N_EDGES + 255) / 256, 256, 0, stream>>>(
      xm, em, ln_g, ln_b, W_sh, b_sh, Se, Ie, vals, deg);
  k_dinv<<<(N_NODES + 255) / 256, 256, 0, stream>>>(deg);

  /* 4. layer 0 */
  k_gemm_l0<<<(NROWS / 16 + 7) / 8, 256, 0, stream>>>((const unsigned int*)Hb, W0t, G0);
  k_scatter<<<(N_EDGES * D_STALK + 255) / 256, 256, 0, stream>>>(
      G0, vals, Se, Ie, deg, agg0, C_MID);
  k_elu_bf<<<(int)(((long)NROWS * C_MID + 255) / 256), 256, 0, stream>>>(
      G0, agg0, b0, H1b, C_MID, (long)NROWS * C_MID);

  /* 5. layer 1 */
  k_gemm_l1<<<(NROWS / 16 + 7) / 8, 256, 0, stream>>>((const unsigned int*)H1b, W1t, G1);
  k_scatter<<<(N_EDGES * D_STALK + 255) / 256, 256, 0, stream>>>(
      G1, vals, Se, Ie, deg, agg1, C_OUT);
  k_elu_f32<<<(int)(((long)NROWS * C_OUT + 255) / 256), 256, 0, stream>>>(
      G1, agg1, b1, out, C_OUT, (long)NROWS * C_OUT);
}
